// UE_GAT_UD_CONCAT_54520314856232
// MI455X (gfx1250) — compile-verified, matrix operand order checked by reading, not run on hardware
//
#include <hip/hip_runtime.h>
#include <stdint.h>

// Problem constants (match the reference)
#define NN 32768
#define EE 131072
#define DD 384
#define HH 8
#define HD (HH * DD)   // 3072
#define D2 (2 * DD)    // 768

typedef __attribute__((ext_vector_type(16))) __bf16 v16bf;
typedef __attribute__((ext_vector_type(8)))  float  v8f;

union Frag {
    v16bf v;
    uint4 q[2];
};

__device__ __forceinline__ unsigned short f2bf(float f) {
    unsigned int u = __float_as_uint(f);
    u += 0x7FFFu + ((u >> 16) & 1u);   // round-to-nearest-even
    return (unsigned short)(u >> 16);
}
// order-preserving encode of float into uint for atomicMax-based segment max
__device__ __forceinline__ unsigned int encf(float x) {
    unsigned int u = __float_as_uint(x);
    return (u & 0x80000000u) ? ~u : (u | 0x80000000u);
}
__device__ __forceinline__ float decf(unsigned int u) {
    u = (u & 0x80000000u) ? (u & 0x7FFFFFFFu) : ~u;
    return __uint_as_float(u);
}
#define ENC_NEG_INF 0x007FFFFFu   // encf(-inf)

// ---------------- CDNA5 async global->LDS copy (ASYNCcnt-tracked, no VGPR round trip)
__device__ __forceinline__ unsigned lds_u32(const void* p) {
    // generic pointer to LDS: low 32 bits are the LDS offset (ISA 10.2 aperture rules)
    return (unsigned)(unsigned long long)p;
}
__device__ __forceinline__ void async_copy_b128(unsigned lds_addr, const void* gptr) {
    asm volatile("global_load_async_to_lds_b128 %0, %1, off"
                 :: "v"(lds_addr), "v"(gptr)
                 : "memory");
}
#if __has_builtin(__builtin_amdgcn_s_wait_asynccnt)
#define ASYNC_WAIT(n) __builtin_amdgcn_s_wait_asynccnt(n)
#else
#define ASYNC_WAIT(n) asm volatile("s_wait_asynccnt %0" :: "i"(n) : "memory")
#endif

// ---------------------------------------------------------------- fills
__global__ void fill_u32_kernel(unsigned int* p, unsigned int v, long n) {
    long i = (long)blockIdx.x * blockDim.x + threadIdx.x;
    if (i < n) p[i] = v;
}
__global__ void fill_f32_kernel(float* p, float v, long n) {
    long i = (long)blockIdx.x * blockDim.x + threadIdx.x;
    if (i < n) p[i] = v;
}

// -------------------------------------------- weight transpose -> bf16
// in: [R][C] f32 row-major  ->  out: [C][R] bf16 row-major
__global__ void transpose_bf16_kernel(const float* __restrict__ in,
                                      unsigned short* __restrict__ out,
                                      int R, int C) {
    __shared__ float tile[32][33];
    int c0 = blockIdx.x * 32, r0 = blockIdx.y * 32;
    for (int j = threadIdx.y; j < 32; j += 8) {
        int r = r0 + j, c = c0 + threadIdx.x;
        tile[j][threadIdx.x] = (r < R && c < C) ? in[(long)r * C + c] : 0.0f;
    }
    __syncthreads();
    for (int j = threadIdx.y; j < 32; j += 8) {
        int c = c0 + j, r = r0 + threadIdx.x;
        if (c < C && r < R) out[(long)c * R + r] = f2bf(tile[threadIdx.x][j]);
    }
}

// ------------------------------------------- concat(f1,f2) -> bf16 [N,768]
__global__ void concat_bf16_kernel(const float* __restrict__ f1,
                                   const float* __restrict__ f2,
                                   unsigned short* __restrict__ xc) {
    long i = (long)blockIdx.x * blockDim.x + threadIdx.x;
    if (i >= (long)NN * D2) return;
    int c = (int)(i % D2);
    long row = i / D2;
    float v = (c < DD) ? f1[row * DD + c] : f2[row * DD + (c - DD)];
    xc[i] = f2bf(v);
}

// ----------------------------------------------------- WMMA bf16 GEMM
// C[M,Nc] (f32 and/or bf16) = act( A[M,K] * B^T  + bias ),  B given as [Nc,K]
// M,Nc multiples of 128; K multiple of 32. 256 threads = 8 waves;
// wave (wm,wn) computes a 32x64 sub-tile = 2x4 wmma 16x16 tiles.
// Global->LDS staging is double-buffered via global_load_async_to_lds_b128.
#define LDSK 40   // padded row stride in shorts (80 B): kills b128 bank conflicts

__global__ __launch_bounds__(256) void gemm_bf16_wmma_kernel(
    const unsigned short* __restrict__ A,   // [M,K] bf16
    const unsigned short* __restrict__ B,   // [Nc,K] bf16 (pre-transposed weights)
    const float* __restrict__ bias,         // [Nc] or null
    float* __restrict__ Cf,                 // [M,Nc] or null
    unsigned short* __restrict__ Cbf,       // [M,Nc] or null
    int M, int Nc, int K, int act)          // act: 0=none 1=relu
{
    __shared__ unsigned short As[2][128 * LDSK];
    __shared__ unsigned short Bs[2][128 * LDSK];

    const int tid  = threadIdx.x;
    const int wave = tid >> 5, lane = tid & 31;
    const int wm = wave >> 1, wn = wave & 1;
    const int l  = lane & 15, hi = lane >> 4;
    const long blockM = (long)blockIdx.y * 128;
    const long blockN = (long)blockIdx.x * 128;

    // staging assignment: thread handles rows r0 and r0+64, 16B column group g
    const int r0 = tid >> 2;        // 0..63
    const int r1 = r0 + 64;         // 64..127
    const int g8 = (tid & 3) * 8;   // column offset in shorts (16B groups)

    const unsigned short* Ar0 = A + (blockM + r0) * (long)K + g8;
    const unsigned short* Ar1 = A + (blockM + r1) * (long)K + g8;
    const unsigned short* Br0 = B + (blockN + r0) * (long)K + g8;
    const unsigned short* Br1 = B + (blockN + r1) * (long)K + g8;

    v8f acc[2][4];
    for (int i = 0; i < 2; ++i)
        for (int j = 0; j < 4; ++j)
            for (int e = 0; e < 8; ++e) acc[i][j][e] = 0.0f;

    const int nk = K >> 5;   // K / 32

    // prologue: async-stage tile 0 into buffer 0 (4 async ops per wave)
    {
        unsigned la = lds_u32(&As[0][r0 * LDSK + g8]);
        unsigned lb = lds_u32(&Bs[0][r0 * LDSK + g8]);
        async_copy_b128(la, Ar0);
        async_copy_b128(la + 64 * LDSK * 2, Ar1);
        async_copy_b128(lb, Br0);
        async_copy_b128(lb + 64 * LDSK * 2, Br1);
    }

    for (int kt = 0; kt < nk; ++kt) {
        const int buf = kt & 1;
        if (kt + 1 < nk) {
            __syncthreads();   // everyone done READING buf^1 (compute of tile kt-1)
            const long k0 = (long)(kt + 1) << 5;
            unsigned la = lds_u32(&As[buf ^ 1][r0 * LDSK + g8]);
            unsigned lb = lds_u32(&Bs[buf ^ 1][r0 * LDSK + g8]);
            async_copy_b128(la, Ar0 + k0);
            async_copy_b128(la + 64 * LDSK * 2, Ar1 + k0);
            async_copy_b128(lb, Br0 + k0);
            async_copy_b128(lb + 64 * LDSK * 2, Br1 + k0);
            ASYNC_WAIT(4);     // retire tile-kt copies (async loads retire in order)
        } else {
            ASYNC_WAIT(0);
        }
        __syncthreads();       // tile kt fully resident in LDS for all waves

        // fragment loads per ISA 16-bit A layout:
        // lanes 0-15 (hi=0): K 0..7 then 16..23 ; lanes 16-31: K 8..15 then 24..31
        Frag a[2], b[4];
        for (int i = 0; i < 2; ++i) {
            const unsigned short* p = &As[buf][(wm * 32 + i * 16 + l) * LDSK + hi * 8];
            a[i].q[0] = *(const uint4*)p;
            a[i].q[1] = *(const uint4*)(p + 16);
        }
        for (int j = 0; j < 4; ++j) {
            const unsigned short* p = &Bs[buf][(wn * 64 + j * 16 + l) * LDSK + hi * 8];
            b[j].q[0] = *(const uint4*)p;
            b[j].q[1] = *(const uint4*)(p + 16);
        }
        for (int i = 0; i < 2; ++i)
            for (int j = 0; j < 4; ++j)
                acc[i][j] = __builtin_amdgcn_wmma_f32_16x16x32_bf16(
                    false, a[i].v, false, b[j].v, (short)0, acc[i][j], false, false);
    }

    // epilogue: C/D layout — VGPR r: M = base + hi*8 + r, N = base + l
    for (int i = 0; i < 2; ++i)
        for (int j = 0; j < 4; ++j) {
            long col = blockN + wn * 64 + j * 16 + l;
            float bv = bias ? bias[col] : 0.0f;
            for (int r = 0; r < 8; ++r) {
                long row = blockM + wm * 32 + i * 16 + hi * 8 + r;
                float v = acc[i][j][r] + bv;
                if (act == 1) v = v > 0.0f ? v : 0.0f;
                long idx = row * (long)Nc + col;
                if (Cf)  Cf[idx]  = v;
                if (Cbf) Cbf[idx] = f2bf(v);
            }
        }
}

// ----------------------------------------- GAT layer-1 edge kernels (H=8)
__global__ __launch_bounds__(256) void edge_logits_h8_kernel(
    const float* __restrict__ fs, const float* __restrict__ fd,
    const int* __restrict__ src, const int* __restrict__ dst,
    const float* __restrict__ attn,        // [8,384]
    float* __restrict__ logits,            // [E,8]
    unsigned int* __restrict__ maxenc) {   // [N,8]
    int e = blockIdx.x;
    int h = threadIdx.x >> 5, lane = threadIdx.x & 31;
    int s = src[e], d = dst[e];
    const float* ps = fs + (long)s * HD + h * DD;
    const float* pd = fd + (long)d * HD + h * DD;
    const float* pa = attn + h * DD;
    float sum = 0.0f;
    for (int i = lane; i < DD; i += 32) {
        float v = ps[i] + pd[i];
        v = v > 0.0f ? v : 0.2f * v;       // leaky_relu
        sum += v * pa[i];
    }
    for (int off = 16; off > 0; off >>= 1) sum += __shfl_down(sum, off, 32);
    if (lane == 0) {
        logits[(long)e * HH + h] = sum;
        atomicMax(&maxenc[(long)d * HH + h], encf(sum));
    }
}

__global__ void edge_exp_h8_kernel(const int* __restrict__ dst,
                                   float* __restrict__ logits,          // in: logit, out: exp
                                   const unsigned int* __restrict__ maxenc,
                                   float* __restrict__ denom) {         // [N,8]
    long i = (long)blockIdx.x * blockDim.x + threadIdx.x;
    if (i >= (long)EE * HH) return;
    int e = (int)(i >> 3), h = (int)(i & 7);
    int d = dst[e];
    float ex = __expf(logits[i] - decf(maxenc[(long)d * HH + h]));
    logits[i] = ex;
    atomicAdd(&denom[(long)d * HH + h], ex);
}

__global__ __launch_bounds__(256) void edge_scatter_h8_kernel(
    const float* __restrict__ fs, const int* __restrict__ src,
    const int* __restrict__ dst, const float* __restrict__ ex,
    const float* __restrict__ denom, float* __restrict__ out) {
    int e = blockIdx.x;
    int h = threadIdx.x >> 5, lane = threadIdx.x & 31;
    int s = src[e], d = dst[e];
    float alpha = ex[(long)e * HH + h] / denom[(long)d * HH + h];
    const float* ps = fs + (long)s * HD + h * DD;
    float* po = out + (long)d * HD + h * DD;
    for (int i = lane; i < DD; i += 32) atomicAdd(&po[i], alpha * ps[i]);
}

__global__ void elu_bias_bf16_kernel(const float* __restrict__ in,
                                     const float* __restrict__ bias,   // [3072]
                                     unsigned short* __restrict__ out) {
    long i = (long)blockIdx.x * blockDim.x + threadIdx.x;
    if (i >= (long)NN * HD) return;
    float v = in[i] + bias[i % HD];
    v = v > 0.0f ? v : (__expf(v) - 1.0f);  // elu
    out[i] = f2bf(v);
}

// ----------------------------------------- GAT layer-2 edge kernels (H=1)
__global__ __launch_bounds__(256) void edge_logits_h1_kernel(
    const float* __restrict__ fs, const float* __restrict__ fd,
    const int* __restrict__ src, const int* __restrict__ dst,
    const float* __restrict__ attn,        // [384]
    float* __restrict__ logits,            // [E]
    unsigned int* __restrict__ maxenc) {   // [N]
    int e = blockIdx.x * 8 + (threadIdx.x >> 5);
    int lane = threadIdx.x & 31;
    if (e >= EE) return;
    int s = src[e], d = dst[e];
    const float* ps = fs + (long)s * DD;
    const float* pd = fd + (long)d * DD;
    float sum = 0.0f;
    for (int i = lane; i < DD; i += 32) {
        float v = ps[i] + pd[i];
        v = v > 0.0f ? v : 0.2f * v;
        sum += v * attn[i];
    }
    for (int off = 16; off > 0; off >>= 1) sum += __shfl_down(sum, off, 32);
    if (lane == 0) {
        logits[e] = sum;
        atomicMax(&maxenc[d], encf(sum));
    }
}

__global__ void edge_exp_h1_kernel(const int* __restrict__ dst,
                                   float* __restrict__ logits,
                                   const unsigned int* __restrict__ maxenc,
                                   float* __restrict__ denom) {
    long i = (long)blockIdx.x * blockDim.x + threadIdx.x;
    if (i >= EE) return;
    int d = dst[i];
    float ex = __expf(logits[i] - decf(maxenc[d]));
    logits[i] = ex;
    atomicAdd(&denom[d], ex);
}

__global__ __launch_bounds__(256) void edge_scatter_h1_kernel(
    const float* __restrict__ fs, const int* __restrict__ src,
    const int* __restrict__ dst, const float* __restrict__ ex,
    const float* __restrict__ denom, float* __restrict__ out) {
    int e = blockIdx.x * 8 + (threadIdx.x >> 5);
    int lane = threadIdx.x & 31;
    if (e >= EE) return;
    int s = src[e], d = dst[e];
    float alpha = ex[e] / denom[d];
    const float* ps = fs + (long)s * DD;
    float* po = out + (long)d * DD;
    for (int i = lane; i < DD; i += 32) atomicAdd(&po[i], alpha * ps[i]);
}

__global__ void final_bias_kernel(const float* __restrict__ in,
                                  const float* __restrict__ bias,  // [384]
                                  float* __restrict__ out) {
    long i = (long)blockIdx.x * blockDim.x + threadIdx.x;
    if (i >= (long)NN * DD) return;
    out[i] = in[i] + bias[i % DD];
}

// ================================================================ launch
static inline long cdiv(long a, long b) { return (a + b - 1) / b; }

extern "C" void kernel_launch(void* const* d_in, const int* in_sizes, int n_in,
                              void* d_out, int out_size, void* d_ws, size_t ws_size,
                              hipStream_t stream) {
    (void)in_sizes; (void)n_in; (void)out_size; (void)ws_size;

    const float* features1 = (const float*)d_in[0];
    const float* features2 = (const float*)d_in[1];
    const int*   src       = (const int*)  d_in[2];
    const int*   dst       = (const int*)  d_in[3];
    const float* Wd        = (const float*)d_in[4];
    const float* bd        = (const float*)d_in[5];
    const float* Wl1       = (const float*)d_in[6];
    const float* Wr1       = (const float*)d_in[7];
    const float* attn1     = (const float*)d_in[8];
    const float* b1        = (const float*)d_in[9];
    const float* Wl2       = (const float*)d_in[10];
    const float* Wr2       = (const float*)d_in[11];
    const float* attn2     = (const float*)d_in[12];
    const float* b2        = (const float*)d_in[13];
    float* out = (float*)d_out;

    // ---- workspace carve-up (256B aligned)
    char* base = (char*)d_ws;
    size_t off = 0;
    auto alloc = [&](size_t bytes) -> char* {
        char* p = base + off;
        off += (bytes + 255) & ~(size_t)255;
        return p;
    };
    unsigned short* WdT  = (unsigned short*)alloc((size_t)DD * D2 * 2);  // [384][768]
    unsigned short* Wl1T = (unsigned short*)alloc((size_t)HD * DD * 2);  // [3072][384]
    unsigned short* Wr1T = (unsigned short*)alloc((size_t)HD * DD * 2);
    unsigned short* Wl2T = (unsigned short*)alloc((size_t)DD * HD * 2);  // [384][3072]
    unsigned short* Wr2T = (unsigned short*)alloc((size_t)DD * HD * 2);
    unsigned short* Xc   = (unsigned short*)alloc((size_t)NN * D2 * 2);  // [N,768] bf16
    unsigned short* Fbf  = (unsigned short*)alloc((size_t)NN * DD * 2);  // feats bf16
    float* fs1   = (float*)alloc((size_t)NN * HD * 4);
    float* fd1   = (float*)alloc((size_t)NN * HD * 4);
    float* out1  = (float*)alloc((size_t)NN * HD * 4);
    unsigned short* Gbf = (unsigned short*)alloc((size_t)NN * HD * 2);   // elu(out1+b1) bf16
    float* fs2   = (float*)alloc((size_t)NN * DD * 4);
    float* fd2   = (float*)alloc((size_t)NN * DD * 4);
    float* lg1   = (float*)alloc((size_t)EE * HH * 4);                   // logits then exp
    unsigned int* mx1 = (unsigned int*)alloc((size_t)NN * HH * 4);
    float* dn1   = (float*)alloc((size_t)NN * HH * 4);
    float* lg2   = (float*)alloc((size_t)EE * 4);
    unsigned int* mx2 = (unsigned int*)alloc((size_t)NN * 4);
    float* dn2   = (float*)alloc((size_t)NN * 4);
    float* out2  = (float*)alloc((size_t)NN * DD * 4);

    dim3 tb(32, 8);
    // ---- weights -> bf16 transposed [Ncols][K]
    transpose_bf16_kernel<<<dim3(DD / 32, D2 / 32), tb, 0, stream>>>(Wd,  WdT,  D2, DD);
    transpose_bf16_kernel<<<dim3(HD / 32, DD / 32), tb, 0, stream>>>(Wl1, Wl1T, DD, HD);
    transpose_bf16_kernel<<<dim3(HD / 32, DD / 32), tb, 0, stream>>>(Wr1, Wr1T, DD, HD);
    transpose_bf16_kernel<<<dim3(DD / 32, HD / 32), tb, 0, stream>>>(Wl2, Wl2T, HD, DD);
    transpose_bf16_kernel<<<dim3(DD / 32, HD / 32), tb, 0, stream>>>(Wr2, Wr2T, HD, DD);

    // ---- concat + convert activations
    concat_bf16_kernel<<<cdiv((long)NN * D2, 256), 256, 0, stream>>>(features1, features2, Xc);

    // ---- dense: feats = relu(Xc @ Wd + bd)  -> bf16
    gemm_bf16_wmma_kernel<<<dim3(DD / 128, NN / 128), 256, 0, stream>>>(
        Xc, WdT, bd, nullptr, Fbf, NN, DD, D2, 1);

    // ---- gat1 projections (fp32 out)
    gemm_bf16_wmma_kernel<<<dim3(HD / 128, NN / 128), 256, 0, stream>>>(
        Fbf, Wl1T, nullptr, fs1, nullptr, NN, HD, DD, 0);
    gemm_bf16_wmma_kernel<<<dim3(HD / 128, NN / 128), 256, 0, stream>>>(
        Fbf, Wr1T, nullptr, fd1, nullptr, NN, HD, DD, 0);

    // ---- gat1 edge softmax + aggregate
    fill_u32_kernel<<<cdiv((long)NN * HH, 256), 256, 0, stream>>>(mx1, ENC_NEG_INF, (long)NN * HH);
    fill_f32_kernel<<<cdiv((long)NN * HH, 256), 256, 0, stream>>>(dn1, 0.0f, (long)NN * HH);
    fill_f32_kernel<<<cdiv((long)NN * HD, 256), 256, 0, stream>>>(out1, 0.0f, (long)NN * HD);
    edge_logits_h8_kernel<<<EE, 256, 0, stream>>>(fs1, fd1, src, dst, attn1, lg1, mx1);
    edge_exp_h8_kernel<<<cdiv((long)EE * HH, 256), 256, 0, stream>>>(dst, lg1, mx1, dn1);
    edge_scatter_h8_kernel<<<EE, 256, 0, stream>>>(fs1, src, dst, lg1, dn1, out1);

    // ---- elu(out1 + b1) -> bf16
    elu_bias_bf16_kernel<<<cdiv((long)NN * HD, 256), 256, 0, stream>>>(out1, b1, Gbf);

    // ---- gat2 projections
    gemm_bf16_wmma_kernel<<<dim3(DD / 128, NN / 128), 256, 0, stream>>>(
        Gbf, Wl2T, nullptr, fs2, nullptr, NN, DD, HD, 0);
    gemm_bf16_wmma_kernel<<<dim3(DD / 128, NN / 128), 256, 0, stream>>>(
        Gbf, Wr2T, nullptr, fd2, nullptr, NN, DD, HD, 0);

    // ---- gat2 edge softmax + aggregate
    fill_u32_kernel<<<cdiv((long)NN, 256), 256, 0, stream>>>(mx2, ENC_NEG_INF, NN);
    fill_f32_kernel<<<cdiv((long)NN, 256), 256, 0, stream>>>(dn2, 0.0f, NN);
    fill_f32_kernel<<<cdiv((long)NN * DD, 256), 256, 0, stream>>>(out2, 0.0f, (long)NN * DD);
    edge_logits_h1_kernel<<<cdiv(EE, 8), 256, 0, stream>>>(fs2, fd2, src, dst, attn2, lg2, mx2);
    edge_exp_h1_kernel<<<cdiv(EE, 256), 256, 0, stream>>>(dst, lg2, mx2, dn2);
    edge_scatter_h1_kernel<<<cdiv(EE, 8), 256, 0, stream>>>(fs2, src, dst, lg2, dn2, out2);

    // ---- + b2 (mean over 1 head == identity)
    final_bias_kernel<<<cdiv((long)NN * DD, 256), 256, 0, stream>>>(out2, b2, out);
}
